// AdditiveAttention_87857851007470
// MI455X (gfx1250) — matrix operand order repaired and checked by reading, not compiled, not run on hardware
//
#include <hip/hip_runtime.h>
#include <hip/hip_bf16.h>
#include <math.h>
#include <stdint.h>

typedef __attribute__((ext_vector_type(16))) __bf16 v16bf;
typedef __attribute__((ext_vector_type(8)))  float  v8f;

#define B_  8
#define Q_  512
#define K_  512
#define H_  64
#define D_  256   // KEY_D == QUERY_D == VAL_D
#define NT_ (K_/32)

// ---------------- Kernel 1: qp = query @ W_q, kp = key @ W_k (fp32) ----------
__global__ void __launch_bounds__(64)
proj_kernel(const float* __restrict__ query, const float* __restrict__ key,
            const float* __restrict__ Wq, const float* __restrict__ Wk,
            float* __restrict__ qp, float* __restrict__ kp)
{
    int row = blockIdx.x;            // 0 .. 2*B*Q-1 (B*Q == B*K)
    int h   = threadIdx.x;           // 0..63
    const float* src; const float* W; float* dst;
    if (row < B_*Q_) {
        src = query + (size_t)row * D_;
        W   = Wq;
        dst = qp + (size_t)row * H_ + h;
    } else {
        int r = row - B_*Q_;
        src = key + (size_t)r * D_;
        W   = Wk;
        dst = kp + (size_t)r * H_ + h;
    }
    float s = 0.f;
#pragma unroll 8
    for (int d = 0; d < D_; ++d)
        s = fmaf(src[d], W[d*H_ + h], s);
    *dst = s;
}

// ---------------- Kernel 2: fused tanh-score + softmax + P@V (WMMA bf16) -----
#define QP_STR 65   // qp rows: kills stride-64 bank conflicts (scalar access)
#define KP_STR 66   // kp rows: 8B-aligned async b64 writes, perfect bank spread
#define P_STR  36   // prob rows: 8B-aligned float2 reads, spread banks
#define V_STRF 36   // value rows (transposed, fp32): 16B-aligned float4 reads

// Async global->LDS copy (CDNA5 ASYNCcnt path). Each lane supplies its own
// LDS byte address (generic-pointer low 32 bits == LDS offset per ISA 10.2)
// and a 32-bit offset against a uniform SGPR base (GVS mode).
__device__ __forceinline__ void async_b32(const void* lds, unsigned voff,
                                          unsigned long long sbase) {
    asm volatile("global_load_async_to_lds_b32 %0, %1, %2"
                 :: "v"((unsigned)(uintptr_t)lds), "v"(voff), "s"(sbase)
                 : "memory");
}
__device__ __forceinline__ void async_b64(const void* lds, unsigned voff,
                                          unsigned long long sbase) {
    asm volatile("global_load_async_to_lds_b64 %0, %1, %2"
                 :: "v"((unsigned)(uintptr_t)lds), "v"(voff), "s"(sbase)
                 : "memory");
}
// 36 async ops per thread per tile are issued; while the next tile is in
// flight we wait for <=36 outstanding (completion is in-order per wave).
__device__ __forceinline__ void async_wait_36() {
    asm volatile("s_wait_asynccnt 0x24" ::: "memory");
}
__device__ __forceinline__ void async_wait_0() {
    asm volatile("s_wait_asynccnt 0x0" ::: "memory");
}

// Issue one K-tile's copies: kp 32x64 f32 (contiguous -> b64), value 32x256
// f32 transposed into LDS (per-element b32 scatter; global side coalesced).
__device__ __forceinline__ void issue_tile_loads(int tid,
                                                 const float* kp_g,
                                                 const float* v_g,
                                                 float* kp_buf, float* v_buf)
{
#pragma unroll
    for (int j = 0; j < 4; ++j) {                 // 1024 float2 chunks / 256 thr
        int idx = tid + j*256;                    // float2 index
        int k = idx >> 5, h2 = idx & 31;
        async_b64(&kp_buf[k*KP_STR + h2*2], (unsigned)(idx*8),
                  (unsigned long long)(uintptr_t)kp_g);
    }
#pragma unroll
    for (int j = 0; j < 32; ++j) {                // 8192 floats / 256 thr
        int e = tid + j*256;
        int k = e >> 8, n = e & 255;
        async_b32(&v_buf[n*V_STRF + k], (unsigned)(e*4),
                  (unsigned long long)(uintptr_t)v_g);
    }
}

// Build a 32x16 bf16 B fragment from one fp32 column of the transposed value
// tile: the lane's 16 K-values are contiguous (two aligned float4 pairs).
__device__ __forceinline__ v16bf load_bcol(const float* c) {
    v16bf r;
#pragma unroll
    for (int j = 0; j < 4; ++j) {
        float4 q = *(const float4*)(c + 4*j);
        r[4*j+0] = (__bf16)q.x; r[4*j+1] = (__bf16)q.y;
        r[4*j+2] = (__bf16)q.z; r[4*j+3] = (__bf16)q.w;
    }
    return r;
}

__global__ void __launch_bounds__(256)
attn_kernel(const float* __restrict__ qp, const float* __restrict__ kp,
            const float* __restrict__ value, const unsigned char* __restrict__ mask,
            const float* __restrict__ w_v, float* __restrict__ out)
{
    __shared__ float qp_s[16*QP_STR];
    __shared__ float kp_s[2][32*KP_STR];          // double-buffered (async dest)
    __shared__ float v_sT[2][D_*V_STRF];          // double-buffered, transposed
    __shared__ float p_s [16*P_STR];
    __shared__ float w_s[H_];
    __shared__ float alpha_s[16], run_max_s[16], run_sum_s[16];

    const int tid  = threadIdx.x;
    const int lane = tid & 31;
    const int wv   = tid >> 5;              // wave 0..7 owns cols [wv*32, wv*32+32)
    const int g    = lane >> 4;             // half-wave (CDNA5 WMMA lane group)
    const int l16  = lane & 15;

    const int b  = blockIdx.x >> 5;         // 32 query tiles per batch
    const int q0 = (blockIdx.x & 31) * 16;

    const float* kp_base = kp    + (size_t)(b*K_)*H_;
    const float* v_base  = value + (size_t)(b*K_)*D_;

    if (tid < H_) w_s[tid] = w_v[tid];
    if (tid < 16) { run_max_s[tid] = -__builtin_inff(); run_sum_s[tid] = 0.f; }
#pragma unroll
    for (int i = 0; i < 4; ++i) {           // 16x64 query-projection tile
        int e = tid + i*256; int m = e >> 6; int h = e & 63;
        qp_s[m*QP_STR + h] = qp[((size_t)(b*Q_ + q0 + m))*H_ + h];
    }

    // prologue: start tile 0's async copies
    issue_tile_loads(tid, kp_base, v_base, kp_s[0], v_sT[0]);

    v8f acc0 = {}; v8f acc1 = {};

    for (int t = 0; t < NT_; ++t) {
        __syncthreads();                    // prev-iter LDS consumers done
        // kick off tile t+1 while we compute tile t
        if (t + 1 < NT_) {
            issue_tile_loads(tid,
                             kp_base + (size_t)(t+1)*32*H_,
                             v_base  + (size_t)(t+1)*32*D_,
                             kp_s[(t+1)&1], v_sT[(t+1)&1]);
            async_wait_36();                // tile t landed (in-order retire)
        } else {
            async_wait_0();
        }
        __syncthreads();                    // tile t visible to all waves

        const float* kpb = kp_s[t&1];
        const float* vb  = v_sT[t&1];
        const int k0 = t*32;

        // -- additive scores: s[m][k] = sum_h w[h]*tanh(q[m][h]+k[k][h]) ---
#pragma unroll
        for (int e = 0; e < 2; ++e) {
            int s = tid*2 + e; int m = s >> 5; int k = s & 31;
            const float* qr = &qp_s[m*QP_STR];
            const float* kr = &kpb[k*KP_STR];
            float sc = 0.f;
#pragma unroll 8
            for (int h = 0; h < H_; ++h)
                sc = fmaf(w_s[h], tanhf(qr[h] + kr[h]), sc);
            bool mk = mask[((size_t)(b*Q_ + q0 + m))*K_ + (k0 + k)] != 0;
            p_s[m*P_STR + k] = mk ? -__builtin_inff() : sc;
        }
        __syncthreads();                    // raw scores ready

        if (tid < 16) {                     // per-row running max + rescale
            float tm = -__builtin_inff();
            for (int k = 0; k < 32; ++k) tm = fmaxf(tm, p_s[tid*P_STR + k]);
            float om = run_max_s[tid];
            float nm = fmaxf(om, tm);
            alpha_s[tid]   = __expf(om - nm);   // exp(-inf)=0 on first tile
            run_max_s[tid] = nm;
        }
        __syncthreads();                    // max/alpha ready

#pragma unroll
        for (int e = 0; e < 2; ++e) {       // exponentiate in place
            int s = tid*2 + e; int m = s >> 5; int k = s & 31;
            p_s[m*P_STR + k] = __expf(p_s[m*P_STR + k] - run_max_s[m]);
        }
        __syncthreads();                    // probabilities ready

        if (tid < 16) {                     // running denominator
            float ts = 0.f;
            for (int k = 0; k < 32; ++k) ts += p_s[tid*P_STR + k];
            run_sum_s[tid] = run_sum_s[tid]*alpha_s[tid] + ts;
        }

        // -- flash-attention correction of the accumulators ----------------
        // C/D layout: VGPR j, lanes 0-15 -> row j; lanes 16-31 -> row j+8
#pragma unroll
        for (int j = 0; j < 8; ++j) {
            float al = alpha_s[j + 8*g];
            acc0[j] *= al; acc1[j] *= al;
        }

        // -- A fragment: 16x32 probs, bf16, contiguous k = 16*g + s (same as B) ----------
        // lane (l16,g): VGPR i holds K-pair 2*i + 16*g
        v16bf a;
#pragma unroll
        for (int i = 0; i < 8; ++i) {
            int kk = 2*i + g*16;
            float2 pv = *(const float2*)&p_s[l16*P_STR + kk];
            a[2*i]   = (__bf16)pv.x;
            a[2*i+1] = (__bf16)pv.y;
        }
        // -- B fragments from the transposed fp32 value tile ---------------
        v16bf b0 = load_bcol(&vb[(wv*32      + l16)*V_STRF + 16*g]);
        v16bf b1 = load_bcol(&vb[(wv*32 + 16 + l16)*V_STRF + 16*g]);

        // one WMMA per N-tile covers the whole K=32 tile
        acc0 = __builtin_amdgcn_wmma_f32_16x16x32_bf16(false, a, false, b0,
                                                       (short)0, acc0, false, false);
        acc1 = __builtin_amdgcn_wmma_f32_16x16x32_bf16(false, a, false, b1,
                                                       (short)0, acc1, false, false);
    }
    __syncthreads();                        // final run_sum visible to all

#pragma unroll
    for (int j = 0; j < 8; ++j) {
        int m = j + 8*g;
        float inv = 1.0f / run_sum_s[m];
        size_t base = ((size_t)(b*Q_ + q0 + m))*D_ + (size_t)wv*32;
        out[base + l16]      = acc0[j] * inv;
        out[base + 16 + l16] = acc1[j] * inv;
    }
}

extern "C" void kernel_launch(void* const* d_in, const int* in_sizes, int n_in,
                              void* d_out, int out_size, void* d_ws, size_t ws_size,
                              hipStream_t stream) {
    // setup_inputs order: key, query, value, mask, W_k, W_q, w_v
    const float* key   = (const float*)d_in[0];
    const float* query = (const float*)d_in[1];
    const float* value = (const float*)d_in[2];
    const unsigned char* mask = (const unsigned char*)d_in[3];
    const float* W_k = (const float*)d_in[4];
    const float* W_q = (const float*)d_in[5];
    const float* w_v = (const float*)d_in[6];
    float* out = (float*)d_out;

    // workspace: qp [B,Q,64] fp32 (1 MiB) + kp [B,K,64] fp32 (1 MiB)
    float* qp = (float*)d_ws;
    float* kp = qp + (size_t)B_*Q_*H_;

    proj_kernel<<<2*B_*Q_, 64, 0, stream>>>(query, key, W_q, W_k, qp, kp);
    attn_kernel<<<B_*(Q_/16), 256, 0, stream>>>(qp, kp, value, mask, w_v, out);

    (void)in_sizes; (void)n_in; (void)out_size; (void)ws_size;
}
